// MaskedMultiHeadAttention_19619410608604
// MI455X (gfx1250) — matrix operand order repaired and checked by reading, not compiled, not run on hardware
//
#include <hip/hip_runtime.h>
#include <hip/hip_bf16.h>
#include <hip/hip_fp16.h>

typedef __attribute__((ext_vector_type(16))) _Float16 v16h;
typedef __attribute__((ext_vector_type(8)))  _Float16 v8h;
typedef __attribute__((ext_vector_type(8)))  float    v8f;
typedef __attribute__((ext_vector_type(4)))  int      v4i;

#define N_HEAD 16
#define SEQ    2048
#define HID    1024
#define HD     64
#define BATCH  4
#define ROWS   (BATCH * SEQ)      // 8192
#define MASKED_BIAS (-10000.0f)
#define SM_SCALE 0.125f           // 1/sqrt(64)

// ---------------- CDNA5 async copy to LDS ----------------
#if __has_builtin(__builtin_amdgcn_global_load_async_to_lds_b128) && \
    __has_builtin(__builtin_amdgcn_s_wait_asynccnt)
#define ASYNC_OK 1
#else
#define ASYNC_OK 0
#endif

#define GAS __attribute__((address_space(1)))
#define LAS __attribute__((address_space(3)))

__device__ __forceinline__ void async_copy16(const _Float16* g, _Float16* l) {
#if ASYNC_OK
  __builtin_amdgcn_global_load_async_to_lds_b128((GAS v4i*)(g), (LAS v4i*)(l), 0, 0);
#else
  *(uint4*)l = *(const uint4*)g;
#endif
}

template <int N>
__device__ __forceinline__ void wait_async() {
#if ASYNC_OK
  __builtin_amdgcn_s_wait_asynccnt(N);
#endif
}

// wave-local LDS store->load fence (Ps is wave-private; DS is in-order per wave)
__device__ __forceinline__ void lds_fence_wave() {
#if __has_builtin(__builtin_amdgcn_s_wait_dscnt)
  __builtin_amdgcn_s_wait_dscnt(0);
#endif
#if __has_builtin(__builtin_amdgcn_wave_barrier)
  __builtin_amdgcn_wave_barrier();
#endif
}

// ---------------- WMMA helpers ----------------
__device__ __forceinline__ v8f wmma16(v16h a, v16h b, v8f c) {
  return __builtin_amdgcn_wmma_f32_16x16x32_f16(
      /*neg_a=*/false, a, /*neg_b=*/false, b,
      /*c_mod=*/(short)0, c, /*reuse_a=*/false, /*reuse_b=*/false);
}

__device__ __forceinline__ v8f vzero8() {
  v8f z = {0.f, 0.f, 0.f, 0.f, 0.f, 0.f, 0.f, 0.f};
  return z;
}

__device__ __forceinline__ v16h cat8(v8h lo, v8h hi) {
  return __builtin_shufflevector(lo, hi, 0, 1, 2, 3, 4, 5, 6, 7,
                                 8, 9, 10, 11, 12, 13, 14, 15);
}

// A fragment (16x32 f16), row-major, pitch in f16 elements (2x 16B loads).
// Lane L: row=L&15, half=L>>4; elems [0..15] = K half*16 + 0..15 (contiguous).
__device__ __forceinline__ v16h load_a_frag(const _Float16* base, int pitch, int lane) {
  const _Float16* p = base + (size_t)(lane & 15) * pitch + ((lane >> 4) << 4);
  return cat8(*(const v8h*)p, *(const v8h*)(p + 8));
}

// B fragment (32x16 f16) from n-major storage: B[k][n] = base[n*pitch + k].
// Lane L: col=L&15, half=L>>4; elems [0..15] = K half*16 + 0..15 (contiguous).
__device__ __forceinline__ v16h load_b_frag(const _Float16* base, int pitch, int lane) {
  const _Float16* p = base + (size_t)(lane & 15) * pitch + ((lane >> 4) << 4);
  return cat8(*(const v8h*)p, *(const v8h*)(p + 8));
}

// ---------------- Kernel 0a: f32 -> f16 (linear) ----------------
__global__ __launch_bounds__(256) void cvt_f32_f16_kernel(
    const float* __restrict__ src, _Float16* __restrict__ dst, int n) {
  int i = (blockIdx.x * 256 + threadIdx.x) * 8;
  if (i >= n) return;
  float4 a = *(const float4*)(src + i);
  float4 b = *(const float4*)(src + i + 4);
  v8h o;
  o[0] = (_Float16)a.x; o[1] = (_Float16)a.y; o[2] = (_Float16)a.z; o[3] = (_Float16)a.w;
  o[4] = (_Float16)b.x; o[5] = (_Float16)b.y; o[6] = (_Float16)b.z; o[7] = (_Float16)b.w;
  *(v8h*)(dst + i) = o;
}

// ---------------- Kernel 0b: f32 -> f16 transpose (W[k][n] -> Wt[n][k]) ----------------
__global__ __launch_bounds__(256) void cvt_transpose_kernel(
    const float* __restrict__ src, _Float16* __restrict__ dst) {
  __shared__ _Float16 tile[32][33];
  const int tx  = threadIdx.x & 31;
  const int ty4 = (threadIdx.x >> 5) * 4;
  const int k0 = blockIdx.x * 32, n0 = blockIdx.y * 32;
#pragma unroll
  for (int i = 0; i < 4; ++i)
    tile[ty4 + i][tx] = (_Float16)src[(size_t)(k0 + ty4 + i) * HID + n0 + tx];
  __syncthreads();
#pragma unroll
  for (int i = 0; i < 4; ++i)
    dst[(size_t)(n0 + ty4 + i) * HID + k0 + tx] = tile[tx][ty4 + i];
}

// ---------------- GEMM tiling ----------------
#define BM 128
#define BN 128
#define BK 32
#define TPITCH (BK + 8)    // 40 f16 = 80B rows (16B aligned); used for A and B tiles
#define NKSTEPS (HID / BK) // 32

// Stage a 128x32 f16 tile (rows strided HID in global): 2 async b128 per thread.
__device__ __forceinline__ void stage_tile_128x32(const _Float16* g,
                                                  _Float16 (*s)[TPITCH], int tid) {
#pragma unroll
  for (int i = 0; i < 2; ++i) {
    int cc  = tid * 2 + i;
    int row = cc >> 2, kc = (cc & 3) * 8;
    async_copy16(g + (size_t)row * HID + kc, &s[row][kc]);
  }
}

// ---------------- Kernel 1: fused QKV projection GEMM ----------------
// grid: (ROWS/128, HID/128, 3); block 256 (8 waves, 2x4, 64x32 per wave)
__global__ __launch_bounds__(256) void qkv_gemm_kernel(
    const _Float16* __restrict__ X16,
    const _Float16* __restrict__ Wqt, const _Float16* __restrict__ Wkt,
    const _Float16* __restrict__ Wvt,
    const float* __restrict__ bq, const float* __restrict__ bk, const float* __restrict__ bv,
    _Float16* __restrict__ q16, _Float16* __restrict__ k16, _Float16* __restrict__ v16t,
    float* __restrict__ presentK, float* __restrict__ presentV) {
  __shared__ _Float16 As[3][BM][TPITCH];
  __shared__ _Float16 Bs[3][BN][TPITCH];

  const int which = blockIdx.z;
  const _Float16* Wt  = (which == 0) ? Wqt : (which == 1) ? Wkt : Wvt;
  const float*   bias = (which == 0) ? bq : (which == 1) ? bk : bv;

  const int m0   = blockIdx.x * BM;
  const int n0   = blockIdx.y * BN;
  const int tid  = threadIdx.x;
  const int lane = tid & 31;
  const int w    = tid >> 5;
  const int wm   = (w >> 2) * 64;
  const int wn   = (w & 3) * 32;

  const _Float16* Abase = X16 + (size_t)m0 * HID;  // + k0
  const _Float16* Bbase = Wt + (size_t)n0 * HID;   // + k0 (n-major)

  v8f acc[4][2];
#pragma unroll
  for (int mi = 0; mi < 4; ++mi)
#pragma unroll
    for (int ni = 0; ni < 2; ++ni) acc[mi][ni] = vzero8();

  stage_tile_128x32(Abase, As[0], tid);
  stage_tile_128x32(Bbase, Bs[0], tid);

  int cur = 0;
  for (int kb = 0; kb < NKSTEPS; ++kb) {
    const int nxt = (cur == 2) ? 0 : cur + 1;
    if (kb + 1 < NKSTEPS) {
      const int k1 = (kb + 1) * BK;
      stage_tile_128x32(Abase + k1, As[nxt], tid);
      stage_tile_128x32(Bbase + k1, Bs[nxt], tid);
      wait_async<4>();
    } else {
      wait_async<0>();
    }
    __syncthreads();

    const _Float16 (*Asc)[TPITCH] = As[cur];
    const _Float16 (*Bsc)[TPITCH] = Bs[cur];
    v16h bfrag0 = load_b_frag(&Bsc[wn][0],      TPITCH, lane);
    v16h bfrag1 = load_b_frag(&Bsc[wn + 16][0], TPITCH, lane);
#pragma unroll
    for (int mi = 0; mi < 4; ++mi) {
      v16h afrag = load_a_frag(&Asc[wm + mi * 16][0], TPITCH, lane);
      acc[mi][0] = wmma16(afrag, bfrag0, acc[mi][0]);
      acc[mi][1] = wmma16(afrag, bfrag1, acc[mi][1]);
    }
    cur = nxt;
  }

  // epilogue
  const int half = lane >> 4, l16 = lane & 15;
#pragma unroll
  for (int mi = 0; mi < 4; ++mi) {
#pragma unroll
    for (int ni = 0; ni < 2; ++ni) {
      const int col  = n0 + wn + ni * 16 + l16;
      const float bb = bias[col];
      const int head = col >> 6, d = col & 63;
      const int rowb = m0 + wm + mi * 16 + half * 8;
      const int b    = rowb >> 11, s_base = rowb & (SEQ - 1);
      const int bhh  = b * N_HEAD + head;
      if (which == 2) {
        // V: transposed f16 layout [bh][d][s] (contiguous in s) + fp32 present
        v8h pk;
#pragma unroll
        for (int v = 0; v < 8; ++v) {
          float val = acc[mi][ni][v] + bb;
          size_t idx = ((size_t)bhh * SEQ + s_base + v) * HD + d;
          presentV[idx] = val;
          pk[v] = (_Float16)val;
        }
        *(v8h*)(v16t + ((size_t)bhh * HD + d) * SEQ + s_base) = pk;
      } else {
        _Float16* dst16 = (which == 0) ? q16 : k16;
#pragma unroll
        for (int v = 0; v < 8; ++v) {
          float val = acc[mi][ni][v] + bb;
          size_t idx = ((size_t)bhh * SEQ + s_base + v) * HD + d;
          dst16[idx] = (_Float16)val;
          if (which == 1) presentK[idx] = val;
        }
      }
    }
  }
}

// ---------------- Kernel 2: causal flash attention ----------------
// grid: (SEQ/128, BATCH*N_HEAD); block 256; wave w owns 16 q-rows.
#define VPITCH 40   // 32 keys + 8 pad
#define PPITCH 40

__device__ __forceinline__ void stage_k_tile(const _Float16* Kg, _Float16 (*Ks)[HD], int tid) {
  int row = tid >> 3, off = (tid & 7) * 8;   // 32 rows x 64 d
  async_copy16(Kg + (size_t)row * HD + off, &Ks[row][off]);
}
__device__ __forceinline__ void stage_vt_tile(const _Float16* Vg, _Float16 (*Vt)[VPITCH], int tid) {
  int row = tid >> 2, off = (tid & 3) * 8;   // 64 d-rows x 32 keys
  async_copy16(Vg + (size_t)row * SEQ + off, &Vt[row][off]);
}

__global__ __launch_bounds__(256) void attn_kernel(
    const _Float16* __restrict__ q16, const _Float16* __restrict__ k16,
    const _Float16* __restrict__ v16t, _Float16* __restrict__ attn16) {
  __shared__ _Float16 Ks[3][32][HD];
  __shared__ _Float16 Vt[3][HD][VPITCH];
  __shared__ _Float16 Ps[8][16][PPITCH];

  const int bh = blockIdx.y;
  const int b  = bh >> 4, h = bh & 15;
  const int q0 = blockIdx.x * 128;
  const int tid = threadIdx.x, lane = tid & 31, w = tid >> 5;
  const int half = lane >> 4, l16 = lane & 15;

  // Q fragments (hd=64 -> 2 frags), contiguous 16B global loads
  const _Float16* Qbase = q16 + ((size_t)bh * SEQ + q0 + w * 16) * HD;
  v16h aq[2];
  aq[0] = load_a_frag(Qbase,      HD, lane);
  aq[1] = load_a_frag(Qbase + 32, HD, lane);

  float m_run[8], l_run[8];
  v8f acc_o[4];
#pragma unroll
  for (int v = 0; v < 8; ++v) { m_run[v] = -1e30f; l_run[v] = 0.f; }
#pragma unroll
  for (int dt = 0; dt < 4; ++dt) acc_o[dt] = vzero8();

  const _Float16* Kbh  = k16 + (size_t)bh * SEQ * HD;
  const _Float16* Vtbh = v16t + (size_t)bh * HD * SEQ;
  const int qrow_base = q0 + w * 16 + half * 8;
  const int NJ = (q0 + 128) / 32;

  stage_k_tile(Kbh, Ks[0], tid);
  stage_vt_tile(Vtbh, Vt[0], tid);

  int cur = 0;
  for (int jb = 0; jb < NJ; ++jb) {
    const int j0 = jb * 32;
    const int nxt = (cur == 2) ? 0 : cur + 1;
    if (jb + 1 < NJ) {
      stage_k_tile(Kbh + (size_t)(j0 + 32) * HD, Ks[nxt], tid);
      stage_vt_tile(Vtbh + (j0 + 32), Vt[nxt], tid);
      wait_async<2>();
    } else {
      wait_async<0>();
    }
    __syncthreads();

    const _Float16 (*Ksc)[HD]     = Ks[cur];
    const _Float16 (*Vtc)[VPITCH] = Vt[cur];

    // S = Q * K^T (16 q-rows x 32 keys)
    v8f sacc[2] = { vzero8(), vzero8() };
#pragma unroll
    for (int nt = 0; nt < 2; ++nt) {
#pragma unroll
      for (int kb = 0; kb < 2; ++kb) {
        v16h bk2 = load_b_frag(&Ksc[nt * 16][kb * 32], HD, lane);
        sacc[nt] = wmma16(aq[kb], bk2, sacc[nt]);
      }
    }

    // online softmax
    float pv[2][8];
#pragma unroll
    for (int v = 0; v < 8; ++v) {
      int qg = qrow_base + v;
      float tmax = -1e30f;
#pragma unroll
      for (int nt = 0; nt < 2; ++nt) {
        int kg = j0 + nt * 16 + l16;
        float s = sacc[nt][v] * SM_SCALE;
        s = (kg > qg) ? MASKED_BIAS : s;
        pv[nt][v] = s;
        tmax = fmaxf(tmax, s);
      }
#pragma unroll
      for (int off = 1; off < 16; off <<= 1) tmax = fmaxf(tmax, __shfl_xor(tmax, off, 32));
      float newm  = fmaxf(m_run[v], tmax);
      float alpha = __expf(m_run[v] - newm);
      m_run[v] = newm;
      float rsum = 0.f;
#pragma unroll
      for (int nt = 0; nt < 2; ++nt) {
        float p = __expf(pv[nt][v] - newm);
        pv[nt][v] = p;
        rsum += p;
      }
#pragma unroll
      for (int off = 1; off < 16; off <<= 1) rsum += __shfl_xor(rsum, off, 32);
      l_run[v] = l_run[v] * alpha + rsum;
#pragma unroll
      for (int dt = 0; dt < 4; ++dt) acc_o[dt][v] *= alpha;
    }

    // transpose P (C-layout f32) -> A-layout f16 via per-wave LDS bounce
#pragma unroll
    for (int nt = 0; nt < 2; ++nt)
#pragma unroll
      for (int v = 0; v < 8; ++v)
        Ps[w][half * 8 + v][nt * 16 + l16] = (_Float16)pv[nt][v];
    lds_fence_wave();

    v16h ap = load_a_frag(&Ps[w][0][0], PPITCH, lane);

    // O += P * V  (B from n-major Vt: B[key][d] = Vt[d][key])
#pragma unroll
    for (int dt = 0; dt < 4; ++dt) {
      v16h bv2 = load_b_frag(&Vtc[dt * 16][0], VPITCH, lane);
      acc_o[dt] = wmma16(ap, bv2, acc_o[dt]);
    }
    cur = nxt;
  }

  // epilogue: O / l, merge heads -> attn16 [B,S,H] f16
  _Float16* outp = attn16 + (size_t)b * SEQ * HID;
#pragma unroll
  for (int dt = 0; dt < 4; ++dt) {
    int col = h * HD + dt * 16 + l16;
#pragma unroll
    for (int v = 0; v < 8; ++v) {
      int qrow = qrow_base + v;
      outp[(size_t)qrow * HID + col] = (_Float16)(acc_o[dt][v] / l_run[v]);
    }
  }
}

// ---------------- Kernel 3: output projection ----------------
__global__ __launch_bounds__(256) void out_proj_kernel(
    const _Float16* __restrict__ A16, const _Float16* __restrict__ Wot,
    const float* __restrict__ bo, float* __restrict__ out) {
  __shared__ _Float16 As[3][BM][TPITCH];
  __shared__ _Float16 Bs[3][BN][TPITCH];

  const int m0   = blockIdx.x * BM;
  const int n0   = blockIdx.y * BN;
  const int tid  = threadIdx.x;
  const int lane = tid & 31;
  const int w    = tid >> 5;
  const int wm   = (w >> 2) * 64;
  const int wn   = (w & 3) * 32;

  const _Float16* Abase = A16 + (size_t)m0 * HID;
  const _Float16* Bbase = Wot + (size_t)n0 * HID;

  v8f acc[4][2];
#pragma unroll
  for (int mi = 0; mi < 4; ++mi)
#pragma unroll
    for (int ni = 0; ni < 2; ++ni) acc[mi][ni] = vzero8();

  stage_tile_128x32(Abase, As[0], tid);
  stage_tile_128x32(Bbase, Bs[0], tid);

  int cur = 0;
  for (int kb = 0; kb < NKSTEPS; ++kb) {
    const int nxt = (cur == 2) ? 0 : cur + 1;
    if (kb + 1 < NKSTEPS) {
      const int k1 = (kb + 1) * BK;
      stage_tile_128x32(Abase + k1, As[nxt], tid);
      stage_tile_128x32(Bbase + k1, Bs[nxt], tid);
      wait_async<4>();
    } else {
      wait_async<0>();
    }
    __syncthreads();

    const _Float16 (*Asc)[TPITCH] = As[cur];
    const _Float16 (*Bsc)[TPITCH] = Bs[cur];
    v16h bfrag0 = load_b_frag(&Bsc[wn][0],      TPITCH, lane);
    v16h bfrag1 = load_b_frag(&Bsc[wn + 16][0], TPITCH, lane);
#pragma unroll
    for (int mi = 0; mi < 4; ++mi) {
      v16h afrag = load_a_frag(&Asc[wm + mi * 16][0], TPITCH, lane);
      acc[mi][0] = wmma16(afrag, bfrag0, acc[mi][0]);
      acc[mi][1] = wmma16(afrag, bfrag1, acc[mi][1]);
    }
    cur = nxt;
  }

  const int half = lane >> 4, l16 = lane & 15;
#pragma unroll
  for (int mi = 0; mi < 4; ++mi) {
#pragma unroll
    for (int ni = 0; ni < 2; ++ni) {
      int col  = n0 + wn + ni * 16 + l16;
      float bb = bo[col];
#pragma unroll
      for (int v = 0; v < 8; ++v) {
        int row = m0 + wm + mi * 16 + half * 8 + v;
        out[(size_t)row * HID + col] = acc[mi][ni][v] + bb;
      }
    }
  }
}

// ---------------- host launch ----------------
extern "C" void kernel_launch(void* const* d_in, const int* in_sizes, int n_in,
                              void* d_out, int out_size, void* d_ws, size_t ws_size,
                              hipStream_t stream) {
  (void)in_sizes; (void)n_in; (void)out_size; (void)ws_size;

  const float* X  = (const float*)d_in[0];
  const float* Wq = (const float*)d_in[1];
  const float* bq = (const float*)d_in[2];
  const float* Wk = (const float*)d_in[3];
  const float* bk = (const float*)d_in[4];
  const float* Wv = (const float*)d_in[5];
  const float* bv = (const float*)d_in[6];
  const float* Wo = (const float*)d_in[7];
  const float* bo = (const float*)d_in[8];

  float* out = (float*)d_out;
  float* presentK = out + (size_t)ROWS * HID;
  float* presentV = presentK + (size_t)ROWS * HID;

  // ws layout (f16): X16 | Wqt | Wkt | Wvt | Wot | q16 | k16 | v16t | attn16
  const size_t XB = (size_t)ROWS * HID * 2;   // 16 MB
  const size_t WB = (size_t)HID * HID * 2;    // 2 MB
  char* ws = (char*)d_ws;
  _Float16* X16    = (_Float16*)(ws);
  _Float16* Wqt    = (_Float16*)(ws + XB);
  _Float16* Wkt    = (_Float16*)(ws + XB + WB);
  _Float16* Wvt    = (_Float16*)(ws + XB + 2 * WB);
  _Float16* Wot    = (_Float16*)(ws + XB + 3 * WB);
  _Float16* q16    = (_Float16*)(ws + XB + 4 * WB);
  _Float16* k16    = (_Float16*)(ws + 2 * XB + 4 * WB);
  _Float16* v16t   = (_Float16*)(ws + 3 * XB + 4 * WB);
  _Float16* attn16 = (_Float16*)(ws + 4 * XB + 4 * WB);

  const int nX = ROWS * HID;
  cvt_f32_f16_kernel<<<nX / (256 * 8), 256, 0, stream>>>(X, X16, nX);
  dim3 gt(HID / 32, HID / 32);
  cvt_transpose_kernel<<<gt, 256, 0, stream>>>(Wq, Wqt);
  cvt_transpose_kernel<<<gt, 256, 0, stream>>>(Wk, Wkt);
  cvt_transpose_kernel<<<gt, 256, 0, stream>>>(Wv, Wvt);
  cvt_transpose_kernel<<<gt, 256, 0, stream>>>(Wo, Wot);

  dim3 g1(ROWS / BM, HID / BN, 3);
  qkv_gemm_kernel<<<g1, 256, 0, stream>>>(X16, Wqt, Wkt, Wvt, bq, bk, bv,
                                          q16, k16, v16t, presentK, presentV);

  dim3 g2(SEQ / 128, BATCH * N_HEAD);
  attn_kernel<<<g2, 256, 0, stream>>>(q16, k16, v16t, attn16);

  dim3 g3(ROWS / BM, HID / BN);
  out_proj_kernel<<<g3, 256, 0, stream>>>(attn16, Wot, bo, out);
}
